// DGRU_23545010717404
// MI455X (gfx1250) — matrix operand order repaired and checked by reading, not compiled, not run on hardware
//
#include <hip/hip_runtime.h>
#include <math.h>

// ---------------- problem constants ----------------
constexpr int B  = 256;
constexpr int T  = 2048;
constexpr int H  = 128;
constexpr int Lyr= 4;
constexpr int G  = 384;          // 3*H gate width
constexpr size_t BT = (size_t)B * T;

typedef __attribute__((ext_vector_type(16))) __bf16 v16bf;
typedef __attribute__((ext_vector_type(8)))  float  v8f;
typedef __attribute__((ext_vector_type(4)))  int    v4i;

// gfx1250 async-copy path (ASYNCcnt), guarded so compile can never break.
#if defined(__gfx1250__) && __has_builtin(__builtin_amdgcn_global_load_async_to_lds_b128) && \
    __has_builtin(__builtin_amdgcn_s_wait_asynccnt)
#define USE_ASYNC_LDS 1
typedef __attribute__((address_space(1))) v4i* gv4i_p;   // global
typedef __attribute__((address_space(3))) v4i* lv4i_p;   // LDS
#else
#define USE_ASYNC_LDS 0
#endif

// ---------------- WMMA fragment helpers ----------------
// B fragment: 32x16 bf16 where B[k][n] = W[n][k], W row-major [n][128].
__device__ inline v16bf ldB_g(const __bf16* p) {
    v16bf b;
    ((uint4*)&b)[0] = *(const uint4*)(p);
    ((uint4*)&b)[1] = *(const uint4*)(p + 16);
    return b;
}
__device__ inline v8f wmma_bf(v16bf a, v16bf b, v8f c) {
    return __builtin_amdgcn_wmma_f32_16x16x32_bf16(false, a, false, b, (short)0, c,
                                                   false, false);
}
__device__ inline v8f v8zero() {
    v8f z = {0.f,0.f,0.f,0.f,0.f,0.f,0.f,0.f};
    return z;
}
__device__ inline float sigm(float x) { return 1.f / (1.f + __expf(-x)); }

// Stage `nchunk` 16-byte chunks from global (row-major bf16) into LDS.
__device__ inline void stage_to_lds(const __bf16* gsrc, __bf16* lds, int nchunk,
                                    int tid, int nthreads) {
#if USE_ASYNC_LDS
    for (int c = tid; c < nchunk; c += nthreads) {
        __builtin_amdgcn_global_load_async_to_lds_b128(
            (gv4i_p)(gsrc + (size_t)c * 8),
            (lv4i_p)(lds + (size_t)c * 8), 0, 0);
    }
    __builtin_amdgcn_s_wait_asynccnt(0);
#else
    for (int c = tid; c < nchunk; c += nthreads) {
        uint4 v = *(const uint4*)(gsrc + (size_t)c * 8);
        *(uint4*)(lds + (size_t)c * 8) = v;
    }
#endif
    __syncthreads();
}

// ---------------- 1) feature extraction ----------------
__global__ void feat_kernel(const float* __restrict__ x, float* __restrict__ feats) {
    size_t bt = (size_t)blockIdx.x * 256 + threadIdx.x;
    float i = x[bt * 2 + 0];
    float q = x[bt * 2 + 1];
    float a2 = i * i + q * q;
    float a  = sqrtf(a2);
    float ia = 1.f / a;
    feats[bt * 6 + 0] = i;
    feats[bt * 6 + 1] = q;
    feats[bt * 6 + 2] = a;
    feats[bt * 6 + 3] = a * a2;
    feats[bt * 6 + 4] = q * ia;
    feats[bt * 6 + 5] = i * ia;
}

// ---------------- 2) fp32 -> bf16 weight conversion ----------------
__global__ void tobf_kernel(const float* __restrict__ s, __bf16* __restrict__ d, int n) {
    int i = blockIdx.x * 256 + threadIdx.x;
    if (i < n) d[i] = (__bf16)s[i];
}

// ---------------- 3) layer-0 input gates (K=6, scalar) ----------------
__global__ void xg0_kernel(const float* __restrict__ feats, const float* __restrict__ W,
                           const float* __restrict__ bias, float* __restrict__ xg) {
    size_t bt = blockIdx.x;
    int n = threadIdx.x;
    float acc = bias[n];
#pragma unroll
    for (int k = 0; k < 6; ++k) acc += feats[bt * 6 + k] * W[n * 6 + k];
    xg[bt * G + n] = acc;
}

// ---------------- 4) big input-gate GEMM, layers 1..3 ----------------
// xg[bt][n] = b_ih[n] + sum_k X[bt][k]*W[n][k];  M=BT, N=384, K=128
// block = 256 (8 waves); WG covers 32 rows; A tile staged once in LDS, then
// all 8 A-fragments preloaded to VGPRs; col-tile outer loop keeps one
// B-fragment live at a time. launch_bounds(.,1) -> full VGPR file, no spills.
__global__ void __launch_bounds__(256, 1)
gemm384_kernel(const __bf16* __restrict__ X, const __bf16* __restrict__ W,
               const float* __restrict__ bias, float* __restrict__ xg) {
    __shared__ __bf16 As[32 * 128];   // 8 KB activation tile
    int tid = threadIdx.x;
    int w = tid >> 5, lane = tid & 31, lm = lane & 15, hi = lane >> 4;
    size_t row0 = (size_t)blockIdx.x * 32;

    // stage A tile: 32 rows x 128 halves = 512 x 16B chunks
    stage_to_lds(X + row0 * 128, As, 512, tid, 256);

    // preload all A fragments (2 row tiles x 4 K-blocks) from LDS
    v16bf aF[2][4];
#pragma unroll
    for (int rt = 0; rt < 2; ++rt)
#pragma unroll
        for (int kb = 0; kb < 4; ++kb) {
            const __bf16* p = &As[(rt * 16 + lm) * 128 + kb * 32 + hi * 8];
            ((uint4*)&aF[rt][kb])[0] = *(const uint4*)(p);
            ((uint4*)&aF[rt][kb])[1] = *(const uint4*)(p + 16);
        }

    v8f c[2][3];
#pragma unroll
    for (int rt = 0; rt < 2; ++rt)
#pragma unroll
        for (int i = 0; i < 3; ++i) c[rt][i] = v8zero();

#pragma unroll
    for (int i = 0; i < 3; ++i) {
        int n0 = (w * 3 + i) * 16;
#pragma unroll
        for (int kb = 0; kb < 4; ++kb) {
            v16bf bf = ldB_g(W + (size_t)(n0 + lm) * 128 + kb * 32 + hi * 8);
            c[0][i] = wmma_bf(aF[0][kb], bf, c[0][i]);
            c[1][i] = wmma_bf(aF[1][kb], bf, c[1][i]);
        }
    }
#pragma unroll
    for (int i = 0; i < 3; ++i) {
        int n = (w * 3 + i) * 16 + lm;
        float bv = bias[n];
#pragma unroll
        for (int rt = 0; rt < 2; ++rt)
#pragma unroll
            for (int vg = 0; vg < 8; ++vg) {
                size_t r = row0 + rt * 16 + vg + 8 * hi;
                xg[r * G + n] = c[rt][i][vg] + bv;
            }
    }
}

// ---------------- 5) sequential GRU recurrence ----------------
// grid = 16 WGs (batch tiles of 16), block = 384 (12 waves).
// Wave w owns gate columns [32w, 32w+32); W_hh fragments resident in VGPRs;
// h state resident in LDS; xg gate rows double-buffered in registers.
__global__ void __launch_bounds__(384, 1)
gru_step_kernel(const float* __restrict__ xg, const __bf16* __restrict__ Whh,
                const float* __restrict__ bhh, const float* __restrict__ h0,
                __bf16* __restrict__ ys) {
    __shared__ __bf16 hbf[16 * 128];
    __shared__ float  hOld[16 * 128];
    __shared__ float  rz[2 * 16 * 128];

    int tid = threadIdx.x;
    int w = tid >> 5, lane = tid & 31, lm = lane & 15, hi = lane >> 4;
    int b0 = blockIdx.x * 16;
    int grp = w >> 2;  // 0:r  1:z  2:n

    // resident W_hh B-fragments: 2 col tiles x 4 K-blocks
    v16bf bW[2][4];
    float biasv[2];
#pragma unroll
    for (int i = 0; i < 2; ++i) {
        int n0 = w * 32 + i * 16;
        biasv[i] = bhh[n0 + lm];
#pragma unroll
        for (int kb = 0; kb < 4; ++kb)
            bW[i][kb] = ldB_g(Whh + (size_t)(n0 + lm) * 128 + kb * 32 + hi * 8);
    }

    // init hidden state from h0
    for (int idx = tid; idx < 16 * 128; idx += 384) {
        int m = idx >> 7, j = idx & 127;
        float hv = h0[(size_t)(b0 + m) * 128 + j];
        hOld[idx] = hv;
        hbf[idx]  = (__bf16)hv;
    }
    __syncthreads();

    auto load_xg = [&](int tt, float (&dst)[2][8]) {
#pragma unroll
        for (int i = 0; i < 2; ++i) {
            int n = w * 32 + i * 16 + lm;
#pragma unroll
            for (int vg = 0; vg < 8; ++vg) {
                int m = vg + 8 * hi;
                dst[i][vg] = xg[((size_t)(b0 + m) * T + tt) * G + n];
            }
        }
    };

    float xcur[2][8], xnxt[2][8];
    load_xg(0, xcur);

    for (int t = 0; t < T; ++t) {
        // hg = h @ W_hh^T for this wave's 32 columns
        v8f acc[2];
        acc[0] = v8zero();
        acc[1] = v8zero();
#pragma unroll
        for (int kb = 0; kb < 4; ++kb) {
            v16bf a;
            const __bf16* p = &hbf[lm * 128 + kb * 32 + hi * 8];
            ((uint4*)&a)[0] = *(const uint4*)(p);
            ((uint4*)&a)[1] = *(const uint4*)(p + 16);
            acc[0] = wmma_bf(a, bW[0][kb], acc[0]);
            acc[1] = wmma_bf(a, bW[1][kb], acc[1]);
        }

        // issue next timestep's gate-row loads (consumed next iteration)
        if (t + 1 < T) load_xg(t + 1, xnxt);
        if (t + 2 < T) {
            __builtin_prefetch(&xg[((size_t)(b0 + lm) * T + t + 2) * G + w * 32], 0, 0);
        }

        float hv[2][8];
#pragma unroll
        for (int i = 0; i < 2; ++i)
#pragma unroll
            for (int vg = 0; vg < 8; ++vg) hv[i][vg] = acc[i][vg] + biasv[i];

        if (grp < 2) {  // r and z gates -> LDS
#pragma unroll
            for (int i = 0; i < 2; ++i) {
                int n = w * 32 + i * 16 + lm;
                int j = n & 127;
#pragma unroll
                for (int vg = 0; vg < 8; ++vg) {
                    int m = vg + 8 * hi;
                    rz[grp * 2048 + m * 128 + j] = sigm(xcur[i][vg] + hv[i][vg]);
                }
            }
        }
        __syncthreads();

        if (grp == 2) {  // n gate + state update
#pragma unroll
            for (int i = 0; i < 2; ++i) {
                int n = w * 32 + i * 16 + lm;
                int j = n & 127;
#pragma unroll
                for (int vg = 0; vg < 8; ++vg) {
                    int m = vg + 8 * hi;
                    int idx = m * 128 + j;
                    float r  = rz[idx];
                    float z  = rz[2048 + idx];
                    float ho = hOld[idx];
                    float nv = tanhf(xcur[i][vg] + r * hv[i][vg]);
                    float hn = (1.f - z) * nv + z * ho;
                    hOld[idx] = hn;
                    hbf[idx]  = (__bf16)hn;
                    ys[((size_t)(b0 + m) * T + t) * 128 + j] = (__bf16)hn;
                }
            }
        }
        __syncthreads();

        // rotate double buffer
#pragma unroll
        for (int i = 0; i < 2; ++i)
#pragma unroll
            for (int vg = 0; vg < 8; ++vg) xcur[i][vg] = xnxt[i][vg];
    }
}

// ---------------- 6) fc_hid: relu(ys3 @ W^T + b), N=K=128 ----------------
__global__ void __launch_bounds__(256, 1)
fchid_kernel(const __bf16* __restrict__ X, const __bf16* __restrict__ W,
             const float* __restrict__ bias, __bf16* __restrict__ hid) {
    __shared__ __bf16 As[16 * 128];   // 4 KB
    int tid = threadIdx.x;
    int w = tid >> 5, lane = tid & 31, lm = lane & 15, hi = lane >> 4;
    size_t row0 = (size_t)blockIdx.x * 16;
    int n0 = w * 16;

    stage_to_lds(X + row0 * 128, As, 256, tid, 256);

    v8f c = v8zero();
#pragma unroll
    for (int kb = 0; kb < 4; ++kb) {
        v16bf a;
        const __bf16* p = &As[lm * 128 + kb * 32 + hi * 8];
        ((uint4*)&a)[0] = *(const uint4*)(p);
        ((uint4*)&a)[1] = *(const uint4*)(p + 16);
        v16bf bf = ldB_g(W + (size_t)(n0 + lm) * 128 + kb * 32 + hi * 8);
        c = wmma_bf(a, bf, c);
    }
    float bv = bias[n0 + lm];
#pragma unroll
    for (int vg = 0; vg < 8; ++vg) {
        float v = c[vg] + bv;
        v = v > 0.f ? v : 0.f;
        hid[(row0 + vg + 8 * hi) * 128 + n0 + lm] = (__bf16)v;
    }
}

// ---------------- 7) final projection (OUT=2, K=134) ----------------
__global__ void out_kernel(const __bf16* __restrict__ hid, const float* __restrict__ feats,
                           const float* __restrict__ Wo, const float* __restrict__ bo,
                           float* __restrict__ out) {
    size_t bt = (size_t)blockIdx.x * 256 + threadIdx.x;
    float a0 = bo[0], a1 = bo[1];
    const __bf16* h = hid + bt * 128;
#pragma unroll 8
    for (int j = 0; j < 128; ++j) {
        float hv = (float)h[j];
        a0 += hv * Wo[j];
        a1 += hv * Wo[134 + j];
    }
#pragma unroll
    for (int k = 0; k < 6; ++k) {
        float f = feats[bt * 6 + k];
        a0 += f * Wo[128 + k];
        a1 += f * Wo[134 + 128 + k];
    }
    out[bt * 2 + 0] = a0;
    out[bt * 2 + 1] = a1;
}

// ---------------- host-side orchestration ----------------
extern "C" void kernel_launch(void* const* d_in, const int* in_sizes, int n_in,
                              void* d_out, int out_size, void* d_ws, size_t ws_size,
                              hipStream_t stream) {
    const float* x  = (const float*)d_in[0];
    const float* h0 = (const float*)d_in[1];
    const float* W_ih[Lyr], *W_hh[Lyr], *b_ih[Lyr], *b_hh[Lyr];
    for (int l = 0; l < Lyr; ++l) {
        W_ih[l] = (const float*)d_in[2 + 4 * l + 0];
        W_hh[l] = (const float*)d_in[2 + 4 * l + 1];
        b_ih[l] = (const float*)d_in[2 + 4 * l + 2];
        b_hh[l] = (const float*)d_in[2 + 4 * l + 3];
    }
    const float* fc_hid_w = (const float*)d_in[18];
    const float* fc_hid_b = (const float*)d_in[19];
    const float* fc_out_w = (const float*)d_in[20];
    const float* fc_out_b = (const float*)d_in[21];
    float* out = (float*)d_out;

    // workspace carve-up
    uint8_t* base = (uint8_t*)d_ws;
    size_t o = 0;
    float*  feats = (float*)(base + o);  o += BT * 6 * sizeof(float);
    float*  xg    = (float*)(base + o);  o += BT * G * sizeof(float);
    __bf16* ysA   = (__bf16*)(base + o); o += BT * H * sizeof(__bf16);
    __bf16* ysB   = (__bf16*)(base + o); o += BT * H * sizeof(__bf16);
    __bf16* hid   = (__bf16*)(base + o); o += BT * H * sizeof(__bf16);
    __bf16* wbf   = (__bf16*)(base + o);
    const int WSZ = G * H;  // 49152 elements per 384x128 matrix
    __bf16* whhbf[Lyr];
    __bf16* wihbf[Lyr];  // only 1..3 used
    for (int l = 0; l < Lyr; ++l) whhbf[l] = wbf + (size_t)l * WSZ;
    for (int l = 1; l < Lyr; ++l) wihbf[l] = wbf + (size_t)(4 + (l - 1)) * WSZ;
    __bf16* fchbf = wbf + (size_t)7 * WSZ;

    // 1) features
    feat_kernel<<<(int)(BT / 256), 256, 0, stream>>>(x, feats);

    // 2) weight conversion to bf16
    for (int l = 0; l < Lyr; ++l)
        tobf_kernel<<<(WSZ + 255) / 256, 256, 0, stream>>>(W_hh[l], whhbf[l], WSZ);
    for (int l = 1; l < Lyr; ++l)
        tobf_kernel<<<(WSZ + 255) / 256, 256, 0, stream>>>(W_ih[l], wihbf[l], WSZ);
    tobf_kernel<<<(H * H + 255) / 256, 256, 0, stream>>>(fc_hid_w, fchbf, H * H);

    // 3) layer 0 input gates (K=6)
    xg0_kernel<<<(int)BT, G, 0, stream>>>(feats, W_ih[0], b_ih[0], xg);

    // 4) layer 0 recurrence
    gru_step_kernel<<<B / 16, 384, 0, stream>>>(xg, whhbf[0], b_hh[0], h0, ysA);

    // 5) layers 1..3: input-gate GEMM + recurrence (ping-pong ys)
    __bf16* prev = ysA;
    __bf16* cur  = ysB;
    for (int l = 1; l < Lyr; ++l) {
        gemm384_kernel<<<(int)(BT / 32), 256, 0, stream>>>(prev, wihbf[l], b_ih[l], xg);
        gru_step_kernel<<<B / 16, 384, 0, stream>>>(xg, whhbf[l], b_hh[l],
                                                    h0 + (size_t)l * B * H, cur);
        __bf16* tmp = prev; prev = cur; cur = tmp;
    }
    // after loop, `prev` holds last layer output

    // 6) fc_hid + relu
    fchid_kernel<<<(int)(BT / 16), 256, 0, stream>>>(prev, fchbf, fc_hid_b, hid);

    // 7) final projection
    out_kernel<<<(int)(BT / 256), 256, 0, stream>>>(hid, feats, fc_out_w, fc_out_b, out);
}
